// W4A16LinearLayer_1623497638587
// MI455X (gfx1250) — matrix operand order repaired and checked by reading, not compiled, not run on hardware
//
#include <hip/hip_runtime.h>

// ---------------------------------------------------------------------------
// W4A16 linear:  out[b,s,o] = sum_i x[b,s,i] * (nib(pw[o,i]) - zp[o]) * sc[o] + bias[o]
// GEMM  M=8192 (B*S), N=11008 (OUT_F), K=4096 (IN_F).
// AI ~1.3 kFLOP/byte -> compute bound -> f16 WMMA (v_wmma_f32_16x16x32_f16),
// fp32 accumulate. Double-buffered LDS pipeline to hide global latency.
// ---------------------------------------------------------------------------

typedef __attribute__((ext_vector_type(16))) _Float16 v16h;
typedef __attribute__((ext_vector_type(8)))  float    v8f;
typedef __attribute__((ext_vector_type(4)))  float    f4;
typedef __attribute__((ext_vector_type(4)))  int      i4;

#define IN_F   4096
#define OUT_F  11008
#define MTOT   8192     // B*S = 4*2048

#define BM 128          // workgroup tile M
#define BN 128          // workgroup tile N
#define BK 64           // K slice per LDS stage (2 WMMA k-steps of 32)
#define LDA 72          // padded K stride (ushort) -> 144B rows, bank-conflict free
#define LDB 72
#define KITERS (IN_F / BK)   // 64

__device__ __forceinline__ unsigned short f2h(float f) {
    _Float16 h = (_Float16)f;
    return __builtin_bit_cast(unsigned short, h);
}

// Per-thread register staging for one K slice (loaded while WMMAs run).
struct AStage { f4 v[2][4]; };   // 32 fp32 activations (two 64-row passes)
struct BStage { i4 v[4];    };   // 16 packed int32 = 32 int4 weights

__device__ __forceinline__ void load_tiles(const float* aptr, const int* bptr, int k0,
                                           AStage& a, BStage& b) {
    #pragma unroll
    for (int p = 0; p < 2; ++p) {
        const f4* s = (const f4*)(aptr + (size_t)p * 64 * IN_F + k0);
        #pragma unroll
        for (int q = 0; q < 4; ++q) a.v[p][q] = s[q];      // 64B contiguous / thread
    }
    const i4* sb = (const i4*)(bptr + k0 / 2);
    #pragma unroll
    for (int q = 0; q < 4; ++q) b.v[q] = sb[q];            // 64B contiguous / thread
}

__device__ __forceinline__ void stage_lds(unsigned short* As, unsigned short* Bs,
                                          const AStage& a, const BStage& b,
                                          int arow, int acolf, int bcol, int bj,
                                          float bscale, float bzp) {
    // A: fp32 -> f16 bits
    #pragma unroll
    for (int p = 0; p < 2; ++p) {
        unsigned short* d = &As[(p * 64 + arow) * LDA + acolf];
        #pragma unroll
        for (int q = 0; q < 4; ++q) {
            f4 v = a.v[p][q];
            d[q * 4 + 0] = f2h(v.x); d[q * 4 + 1] = f2h(v.y);
            d[q * 4 + 2] = f2h(v.z); d[q * 4 + 3] = f2h(v.w);
        }
    }
    // B: unpack nibbles (low = even K, high = odd K, matches reference), dequant
    #pragma unroll
    for (int q = 0; q < 4; ++q) {
        i4 v = b.v[q];
        int e[4] = {v.x, v.y, v.z, v.w};
        #pragma unroll
        for (int ei = 0; ei < 4; ++ei) {
            unsigned short* d = &Bs[bcol * LDB + (bj + q * 4 + ei) * 2];
            d[0] = f2h(((float)(e[ei] & 15)        - bzp) * bscale);
            d[1] = f2h(((float)((e[ei] >> 4) & 15) - bzp) * bscale);
        }
    }
}

__device__ __forceinline__ void wmma_tiles(const unsigned short* As, const unsigned short* Bs,
                                           v8f acc[4][2], int lane, int waveM, int waveN) {
    #pragma unroll
    for (int ks = 0; ks < 2; ++ks) {
        const int kb = ks * 32;
        // A fragments (ISA 16-bit A layout): lane L -> row M=L&15; upper half-wave
        // takes K-pairs offset by 8; VGPR j holds K = (j<4 ? 2j : 16+2(j-4)).
        union { v16h v; unsigned int u[8]; } afrag[4];
        const int mrow  = lane & 15;
        const int koffA = kb + ((lane >> 4) ? 8 : 0);
        #pragma unroll
        for (int i = 0; i < 4; ++i) {
            const unsigned short* ap = &As[(waveM * 64 + i * 16 + mrow) * LDA];
            #pragma unroll
            for (int j = 0; j < 8; ++j) {
                const int kk = koffA + ((j < 4) ? 2 * j : 16 + 2 * (j - 4));
                afrag[i].u[j] = *(const unsigned int*)(ap + kk);
            }
        }
        // B fragments: lane L -> column N=L&15; same K order as A (upper half-wave
        // offset by 8; VGPR u holds K = (u<4 ? 2u : 16+2(u-4))), [n][k] layout.
        union { v16h v; unsigned int u[8]; } bfrag[2];
        const int koffB = kb + ((lane >> 4) ? 8 : 0);
        #pragma unroll
        for (int j = 0; j < 2; ++j) {
            const unsigned short* bp =
                &Bs[(waveN * 32 + j * 16 + (lane & 15)) * LDB + koffB];
            #pragma unroll
            for (int u = 0; u < 8; ++u)
                bfrag[j].u[u] = *(const unsigned int*)(bp + ((u < 4) ? 2 * u : 16 + 2 * (u - 4)));
        }
        #pragma unroll
        for (int i = 0; i < 4; ++i)
            #pragma unroll
            for (int j = 0; j < 2; ++j)
                acc[i][j] = __builtin_amdgcn_wmma_f32_16x16x32_f16(
                    /*neg_a=*/false, afrag[i].v,
                    /*neg_b=*/false, bfrag[j].v,
                    /*c_mod=*/(short)0, acc[i][j],
                    /*reuse_a=*/false, /*reuse_b=*/false);
    }
}

__global__ __launch_bounds__(256)
void w4a16_wmma_kernel(const float* __restrict__ x,
                       const int*   __restrict__ pw,      // [OUT_F][IN_F/2]
                       const float* __restrict__ scales,  // [OUT_F]
                       const int*   __restrict__ zps,     // [OUT_F]
                       const float* __restrict__ bias,    // [OUT_F]
                       float*       __restrict__ out) {   // [MTOT][OUT_F]
    // Double-buffered tiles: 2 * (18 KB + 18 KB) = 72 KB of the WGP's 320 KB LDS.
    __shared__ unsigned short Asmem[2][BM * LDA];
    __shared__ unsigned short Bsmem[2][BN * LDB];

    const int t     = threadIdx.x;
    const int lane  = t & 31;
    const int wave  = t >> 5;
    const int waveM = wave & 1;      // 2 waves along M -> 2*64 = 128
    const int waveN = wave >> 1;     // 4 waves along N -> 4*32 = 128

    const int blockN = blockIdx.x * BN;
    const int blockM = blockIdx.y * BM;

    v8f acc[4][2];
    #pragma unroll
    for (int i = 0; i < 4; ++i)
        #pragma unroll
        for (int j = 0; j < 2; ++j)
            acc[i][j] = (v8f){0.f, 0.f, 0.f, 0.f, 0.f, 0.f, 0.f, 0.f};

    // Global-load mappings (fixed across the K loop)
    const int arow  = t >> 2;            // 0..63 (+64 second pass)
    const int acolf = (t & 3) * 16;      // 16 contiguous floats per thread per row-pass
    const int bcol  = t >> 1;            // 0..127
    const int bj    = (t & 1) * 16;      // 16 contiguous int32 per thread

    const float* aptr = x  + (size_t)(blockM + arow) * IN_F + acolf;
    const int*   bptr = pw + (size_t)(blockN + bcol) * (IN_F / 2) + bj;
    const float bscale = scales[blockN + bcol];
    const float bzp    = (float)zps[blockN + bcol];

    AStage a; BStage b;

    // Prologue: stage slice 0 into buffer 0.
    load_tiles(aptr, bptr, 0, a, b);
    stage_lds(&Asmem[0][0], &Bsmem[0][0], a, b, arow, acolf, bcol, bj, bscale, bzp);

    // Pipelined main loop: global loads for kt+1 overlap WMMAs on kt.
    for (int kt = 0; kt < KITERS - 1; ++kt) {
        const int cur = kt & 1;
        __syncthreads();                              // buf[cur] staged, buf[cur^1] free

        load_tiles(aptr, bptr, (kt + 1) * BK, a, b);  // LOADcnt in flight during WMMAs

        // Speculative prefetch two slices ahead (dropped silently past the end).
        __builtin_prefetch(aptr + (kt + 2) * BK, 0, 3);
        __builtin_prefetch(bptr + (kt + 2) * (BK / 2), 0, 3);

        wmma_tiles(&Asmem[cur][0], &Bsmem[cur][0], acc, lane, waveM, waveN);

        stage_lds(&Asmem[cur ^ 1][0], &Bsmem[cur ^ 1][0], a, b,
                  arow, acolf, bcol, bj, bscale, bzp);
    }
    __syncthreads();
    wmma_tiles(&Asmem[(KITERS - 1) & 1][0], &Bsmem[(KITERS - 1) & 1][0],
               acc, lane, waveM, waveN);

    // Epilogue: C layout (VGPR r -> M = r + 8*(lane>=16), N = lane&15) + bias.
    const int nlane = lane & 15;
    const int mhalf = (lane >> 4) * 8;
    #pragma unroll
    for (int j = 0; j < 2; ++j) {
        const int col = blockN + waveN * 32 + j * 16 + nlane;
        const float bv = bias[col];
        #pragma unroll
        for (int i = 0; i < 4; ++i) {
            const int rbase = blockM + waveM * 64 + i * 16 + mhalf;
            #pragma unroll
            for (int r = 0; r < 8; ++r)
                out[(size_t)(rbase + r) * OUT_F + col] = acc[i][j][r] + bv;
        }
    }
}

extern "C" void kernel_launch(void* const* d_in, const int* in_sizes, int n_in,
                              void* d_out, int out_size, void* d_ws, size_t ws_size,
                              hipStream_t stream) {
    const float* x      = (const float*)d_in[0];
    const int*   pw     = (const int*)  d_in[1];
    const float* scales = (const float*)d_in[2];
    const int*   zps    = (const int*)  d_in[3];
    const float* bias   = (const float*)d_in[4];
    float*       out    = (float*)d_out;

    dim3 grid(OUT_F / BN, MTOT / BM);   // 86 x 64 workgroups
    dim3 block(256);                    // 8 wave32s
    hipLaunchKernelGGL(w4a16_wmma_kernel, grid, block, 0, stream,
                       x, pw, scales, zps, bias, out);
}
